// MODEL_50019189129644
// MI455X (gfx1250) — matrix operand order repaired and checked by reading, not compiled, not run on hardware
//
#include <hip/hip_runtime.h>
#include <hip/hip_bf16.h>
#include <math.h>

typedef __attribute__((ext_vector_type(16))) _Float16 v16h;
typedef __attribute__((ext_vector_type(8)))  _Float16 v8h;
typedef __attribute__((ext_vector_type(8)))  float    v8f;

#define BATCH 8
#define HW 4096   // 64*64

// ---------------- weight norm (optionally reorder K to kk*Cin+ci) ----------------
__global__ void k_wnorm(const float* __restrict__ v, const float* __restrict__ g,
                        float* __restrict__ out, int K, int reorder, int Cin) {
    int o = blockIdx.x;
    const float* vr = v + (size_t)o * K;
    __shared__ float sm[256];
    float s = 0.f;
    for (int k = threadIdx.x; k < K; k += 256) { float a = vr[k]; s += a * a; }
    sm[threadIdx.x] = s;
    __syncthreads();
    for (int off = 128; off > 0; off >>= 1) {
        if (threadIdx.x < off) sm[threadIdx.x] += sm[threadIdx.x + off];
        __syncthreads();
    }
    float scale = g[o] / sqrtf(sm[0]);
    float* orow = out + (size_t)o * K;
    for (int k = threadIdx.x; k < K; k += 256) {
        float w = vr[k] * scale;
        if (reorder) { int ci = k / 9, kk = k - ci * 9; orow[kk * Cin + ci] = w; }
        else orow[k] = w;
    }
}

// ---------------- f32 -> f16 with M padding (for tail weights) ----------------
__global__ void k_cast16(const float* __restrict__ in, _Float16* __restrict__ out,
                         int O, int K, int total) {
    int idx = blockIdx.x * 256 + threadIdx.x;
    if (idx >= total) return;
    int k = idx % K, o = idx / K;
    out[idx] = (o < O) ? (_Float16)in[(size_t)o * K + k] : (_Float16)0.f;
}

// ---------------- head conv 3->64, 3x3 pad1, subtract mean ----------------
__global__ void k_head(const float* __restrict__ x, const float* __restrict__ w,
                       const float* __restrict__ bias, float* __restrict__ h, int total) {
    int idx = blockIdx.x * 256 + threadIdx.x;
    if (idx >= total) return;
    int px = idx & (HW - 1); int o = (idx >> 12) & 63; int b = idx >> 18;
    int y = px >> 6, xx = px & 63;
    float acc = bias[o];
    const float* wr = w + o * 27;
    for (int ci = 0; ci < 3; ++ci)
        for (int ky = 0; ky < 3; ++ky) {
            int yy = y + ky - 1; if ((unsigned)yy > 63u) continue;
            for (int kx = 0; kx < 3; ++kx) {
                int xc = xx + kx - 1; if ((unsigned)xc > 63u) continue;
                acc += (x[(((size_t)b * 3 + ci) << 12) + (yy << 6) + xc] - 0.5f)
                       * wr[ci * 9 + ky * 3 + kx];
            }
        }
    h[idx] = acc;
}

// ---------------- skip conv 3->12, 5x5 pad2, subtract mean ----------------
__global__ void k_skip(const float* __restrict__ x, const float* __restrict__ w,
                       const float* __restrict__ bias, float* __restrict__ out, int total) {
    int idx = blockIdx.x * 256 + threadIdx.x;
    if (idx >= total) return;
    int px = idx & (HW - 1); int r = idx >> 12; int o = r % 12; int b = r / 12;
    int y = px >> 6, xx = px & 63;
    float acc = bias[o];
    const float* wr = w + o * 75;
    for (int ci = 0; ci < 3; ++ci)
        for (int ky = 0; ky < 5; ++ky) {
            int yy = y + ky - 2; if ((unsigned)yy > 63u) continue;
            for (int kx = 0; kx < 5; ++kx) {
                int xc = xx + kx - 2; if ((unsigned)xc > 63u) continue;
                acc += (x[(((size_t)b * 3 + ci) << 12) + (yy << 6) + xc] - 0.5f)
                       * wr[ci * 25 + ky * 5 + kx];
            }
        }
    out[idx] = acc;
}

// ---------------- NCHW f32 -> NHWC-padded(66x66) f16 ----------------
__global__ void k_pad(const float* __restrict__ act, _Float16* __restrict__ out,
                      int C, int total) {
    int idx = blockIdx.x * 256 + threadIdx.x;
    if (idx >= total) return;
    int c = idx % C; int r = idx / C;
    int x = r % 66; r /= 66; int y = r % 66; int b = r / 66;
    float v = 0.f;
    if (y >= 1 && y <= 64 && x >= 1 && x <= 64)
        v = act[(((size_t)b * C + c) << 12) + ((y - 1) << 6) + (x - 1)];
    out[idx] = (_Float16)v;
}

// ---------------- routing: channel means -> logits -> softmax over G=2 ----------------
__global__ void k_route(const float* __restrict__ act, int C,
                        const float* __restrict__ mw, const float* __restrict__ mb,
                        float* __restrict__ s) {
    int b = blockIdx.x;
    __shared__ float mean[256];
    __shared__ float lg[8];
    int t = threadIdx.x;
    if (t < C) {
        const float* ap = act + (((size_t)b * C + t) << 12);
        float su = 0.f;
        for (int i = 0; i < HW; ++i) su += ap[i];
        mean[t] = su * (1.f / 4096.f);
    }
    __syncthreads();
    if (t < 8) {
        float l = mb[t];
        const float* mwr = mw + t * C;
        for (int c = 0; c < C; ++c) l += mean[c] * mwr[c];
        lg[t] = l;
    }
    __syncthreads();
    if (t < 8) {
        int c = t & 3;
        float l0 = lg[c], l1 = lg[4 + c];
        float m = fmaxf(l0, l1);
        float e0 = expf(l0 - m), e1 = expf(l1 - m);
        s[b * 8 + t] = ((t < 4) ? e0 : e1) / (e0 + e1);
    }
}

// ---------------- per-sample expert mixing into f16 weights + f32 bias ----------------
// wn rows: (g*4 + c)*D + d ; mixed rows: c*D + d ; s[b*8 + g*4 + c]
__global__ void k_mix(const float* __restrict__ wn, const float* __restrict__ bias,
                      const float* __restrict__ s, _Float16* __restrict__ wmix,
                      float* __restrict__ bmix, int D, int K, int total) {
    int idx = blockIdx.x * 256 + threadIdx.x;
    if (idx >= total) return;
    int CD = 4 * D;
    int k = idx % K; int r = idx / K; int oc = r % CD; int b = r / CD;
    int c = oc / D, d = oc - c * D;
    float s0 = s[b * 8 + c], s1 = s[b * 8 + 4 + c];
    size_t r0 = (size_t)(c * D + d) * K;
    size_t r1 = (size_t)((4 + c) * D + d) * K;
    wmix[(size_t)idx] = (_Float16)(s0 * wn[r0 + k] + s1 * wn[r1 + k]);
    if (k == 0)
        bmix[b * CD + oc] = s0 * bias[c * D + d] + s1 * bias[(4 + c) * D + d];
}

// ---------------- implicit-GEMM conv via WMMA 16x16x32 f16 ----------------
// One wave computes a 16(M) x 64(N) strip: 4 N-tiles sharing one image row y,
// reusing the A fragment across all 4 WMMAs per k-step. K ordering kk*CIN+ci;
// loop restructured (ky,kx outer / ci inner) => no division, constant strides.
// inpad: [B][66][66][CIN] f16 (zero-padded)
// W:     [B?][Mtot][K] f16 rows (wstride elements per sample; 0 => shared)
// out:   [B][Cout][64][64] f32 ; optional residual add, per-sample bias, relu
template<int CIN>
__global__ void __launch_bounds__(32) k_conv_wmma(
    const _Float16* __restrict__ inpad, const _Float16* __restrict__ W, size_t wstride,
    const float* __restrict__ bias, int bstride,
    float* __restrict__ out, const float* __restrict__ residual,
    int Cout, int relu) {
#if defined(__HIP_DEVICE_COMPILE__)
    constexpr int K = CIN * 9;
    const int lane = threadIdx.x;
    const int nlo = lane & 15, hi = lane >> 4;
    const int y  = blockIdx.x;          // image row: 64 columns per wave
    const int m0 = blockIdx.y << 4;
    const int b  = blockIdx.z;
    // A row: M = m0 + (lane&15); hi lanes read K-offsets +8 per ISA layout
    const _Float16* wrow = W + (size_t)b * wstride + (size_t)(m0 + nlo) * K + (hi ? 8 : 0);
    v8f acc[4] = {};
    for (int ky = 0; ky < 3; ++ky) {
        const _Float16* prow = inpad + (((size_t)b * 66 + (y + ky)) * 66) * CIN + (hi << 3);
        for (int kx = 0; kx < 3; ++kx) {
            const _Float16* rowbase = prow + (size_t)kx * CIN;   // column offset added per tile
            const _Float16* wbase   = wrow + (ky * 3 + kx) * CIN;
            for (int cb = 0; cb < CIN; cb += 32) {
                v8h alo = *(const v8h*)(wbase + cb);
                v8h ahi = *(const v8h*)(wbase + cb + 16);
                v16h a;
#pragma unroll
                for (int i = 0; i < 8; ++i) { a[i] = alo[i]; a[i + 8] = ahi[i]; }
#pragma unroll
                for (int t = 0; t < 4; ++t) {
                    const _Float16* bp = rowbase + (size_t)(t * 16 + nlo) * CIN + cb;
                    v8h blo = *(const v8h*)bp;
                    v8h bhi = *(const v8h*)(bp + 16);
                    v16h bb;
#pragma unroll
                    for (int i = 0; i < 8; ++i) { bb[i] = blo[i]; bb[i + 8] = bhi[i]; }
                    acc[t] = __builtin_amdgcn_wmma_f32_16x16x32_f16(
                        false, a, false, bb, (short)0, acc[t], false, false);
                }
            }
        }
    }
#pragma unroll
    for (int t = 0; t < 4; ++t) {
        const int xx = t * 16 + nlo;
#pragma unroll
        for (int r = 0; r < 8; ++r) {
            int M = m0 + r + (hi << 3);            // C/D layout: M = r + 8*(lane>=16)
            if (M < Cout) {
                size_t oidx = (((size_t)b * Cout + M) << 12) + (size_t)(y << 6) + xx;
                float val = acc[t][r] + bias[(size_t)b * bstride + M];
                if (residual) val += residual[oidx];
                if (relu) val = fmaxf(val, 0.f);
                out[oidx] = val;
            }
        }
    }
#endif
}

// ---------------- pixel shuffle r=2 + skip + mean ----------------
__global__ void k_final(const float* __restrict__ body, const float* __restrict__ skip,
                        float* __restrict__ outp, int total) {
    int idx = blockIdx.x * 256 + threadIdx.x;
    if (idx >= total) return;
    int X = idx & 127; int t = idx >> 7; int Y = t & 127; t >>= 7;
    int c = t % 3; int b = t / 3;
    int ch = c * 4 + (Y & 1) * 2 + (X & 1);
    size_t src = (((size_t)b * 12 + ch) << 12) + ((Y >> 1) << 6) + (X >> 1);
    outp[idx] = body[src] + skip[src] + 0.5f;
}

extern "C" void kernel_launch(void* const* d_in, const int* in_sizes, int n_in,
                              void* d_out, int out_size, void* d_ws, size_t ws_size,
                              hipStream_t stream) {
    (void)in_sizes; (void)n_in; (void)out_size; (void)ws_size;
    const float* x      = (const float*)d_in[0];
    const float* head_v = (const float*)d_in[1];
    const float* head_g = (const float*)d_in[2];
    const float* head_b = (const float*)d_in[3];
    const float* b1_v   = (const float*)d_in[4];
    const float* b1_g   = (const float*)d_in[5];
    const float* b1_b   = (const float*)d_in[6];
    const float* b1_mw  = (const float*)d_in[7];
    const float* b1_mb  = (const float*)d_in[8];
    const float* b2_v   = (const float*)d_in[9];
    const float* b2_g   = (const float*)d_in[10];
    const float* b2_b   = (const float*)d_in[11];
    const float* b2_mw  = (const float*)d_in[12];
    const float* b2_mb  = (const float*)d_in[13];
    const float* tail_v = (const float*)d_in[14];
    const float* tail_g = (const float*)d_in[15];
    const float* tail_b = (const float*)d_in[16];
    const float* skip_v = (const float*)d_in[17];
    const float* skip_g = (const float*)d_in[18];
    const float* skip_b = (const float*)d_in[19];

    // ---- workspace carve-up (256B aligned) ----
    char* p = (char*)d_ws;
    auto alloc = [&](size_t bytes) { void* r = (void*)p; p += (bytes + 255) & ~(size_t)255; return r; };
    float* wn_head = (float*)alloc(64 * 27 * 4);
    float* wn_tail = (float*)alloc(12 * 576 * 4);
    float* wn_skip = (float*)alloc(12 * 75 * 4);
    float* wn_b1   = (float*)alloc((size_t)2048 * 576 * 4);
    float* wn_b2   = (float*)alloc((size_t)512 * 2304 * 4);
    float* h_buf   = (float*)alloc((size_t)BATCH * 64 * HW * 4);
    float* t_buf   = (float*)alloc((size_t)BATCH * 256 * HW * 4);
    float* body    = (float*)alloc((size_t)BATCH * 12 * HW * 4);
    float* skipout = (float*)alloc((size_t)BATCH * 12 * HW * 4);
    float* bmix1   = (float*)alloc(BATCH * 256 * 4);
    float* bmix2   = (float*)alloc(BATCH * 64 * 4);
    float* sbuf    = (float*)alloc(BATCH * 8 * 4);
    _Float16* hpad  = (_Float16*)alloc((size_t)BATCH * 66 * 66 * 64 * 2);
    _Float16* tpad  = (_Float16*)alloc((size_t)BATCH * 66 * 66 * 256 * 2);
    _Float16* wmix1 = (_Float16*)alloc((size_t)BATCH * 256 * 576 * 2);
    _Float16* wmix2 = (_Float16*)alloc((size_t)BATCH * 64 * 2304 * 2);
    _Float16* tailw = (_Float16*)alloc(16 * 576 * 2);

    // ---- static weight prep (no activation dependence) ----
    k_wnorm<<<64,   256, 0, stream>>>(head_v, head_g, wn_head, 27,   0, 0);
    k_wnorm<<<2048, 256, 0, stream>>>(b1_v,   b1_g,   wn_b1,   576,  1, 64);
    k_wnorm<<<512,  256, 0, stream>>>(b2_v,   b2_g,   wn_b2,   2304, 1, 256);
    k_wnorm<<<12,   256, 0, stream>>>(tail_v, tail_g, wn_tail, 576,  1, 64);
    k_wnorm<<<12,   256, 0, stream>>>(skip_v, skip_g, wn_skip, 75,   0, 0);
    k_cast16<<<(16 * 576 + 255) / 256, 256, 0, stream>>>(wn_tail, tailw, 12, 576, 16 * 576);

    // ---- head ----
    int tot_h = BATCH * 64 * HW;
    k_head<<<tot_h / 256, 256, 0, stream>>>(x, wn_head, head_b, h_buf, tot_h);

    // ---- residual blocks ----
    const int padh_tot = BATCH * 66 * 66 * 64;
    const int padt_tot = BATCH * 66 * 66 * 256;
    const int mix1_tot = BATCH * 256 * 576;
    const int mix2_tot = BATCH * 64 * 2304;
    for (int l = 0; l < 4; ++l) {
        k_pad<<<(padh_tot + 255) / 256, 256, 0, stream>>>(h_buf, hpad, 64, padh_tot);
        k_route<<<BATCH, 256, 0, stream>>>(h_buf, 64, b1_mw + (size_t)l * 8 * 64, b1_mb + l * 8, sbuf);
        k_mix<<<(mix1_tot + 255) / 256, 256, 0, stream>>>(
            wn_b1 + (size_t)l * 512 * 576, b1_b + l * 512, sbuf, wmix1, bmix1, 64, 576, mix1_tot);
        dim3 g1(64, 16, BATCH);   // 64 image rows x 16 M-tiles x batch
        k_conv_wmma<64><<<g1, 32, 0, stream>>>(hpad, wmix1, (size_t)256 * 576, bmix1, 256,
                                               t_buf, nullptr, 256, 1);
        k_pad<<<(padt_tot + 255) / 256, 256, 0, stream>>>(t_buf, tpad, 256, padt_tot);
        k_route<<<BATCH, 256, 0, stream>>>(t_buf, 256, b2_mw + (size_t)l * 8 * 256, b2_mb + l * 8, sbuf);
        k_mix<<<(mix2_tot + 255) / 256, 256, 0, stream>>>(
            wn_b2 + (size_t)l * 128 * 2304, b2_b + l * 128, sbuf, wmix2, bmix2, 16, 2304, mix2_tot);
        dim3 g2(64, 4, BATCH);
        k_conv_wmma<256><<<g2, 32, 0, stream>>>(tpad, wmix2, (size_t)64 * 2304, bmix2, 64,
                                                h_buf, h_buf, 64, 0);
    }

    // ---- tail (WMMA, M padded to 16) ----
    k_pad<<<(padh_tot + 255) / 256, 256, 0, stream>>>(h_buf, hpad, 64, padh_tot);
    dim3 g3(64, 1, BATCH);
    k_conv_wmma<64><<<g3, 32, 0, stream>>>(hpad, tailw, 0, tail_b, 0, body, nullptr, 12, 0);

    // ---- skip + pixel shuffle ----
    int tot_s = BATCH * 12 * HW;
    k_skip<<<(tot_s + 255) / 256, 256, 0, stream>>>(x, wn_skip, skip_b, skipout, tot_s);
    int tot_o = BATCH * 3 * 128 * 128;
    k_final<<<(tot_o + 255) / 256, 256, 0, stream>>>(body, skipout, (float*)d_out, tot_o);
}
